// Block_15857019256918
// MI455X (gfx1250) — compile-verified, matrix operand order checked by reading, not run on hardware
//
#include <hip/hip_runtime.h>
#include <math.h>

typedef __attribute__((ext_vector_type(16))) __bf16 v16bf;
typedef __attribute__((ext_vector_type(8)))  float  v8f;

#define DIM   768
#define NHEAD 12
#define HD    64
#define WSZ   14
#define NWIN  100           // 4 * 5 * 5 windows
#define NTOK  196           // 14*14 tokens / window
#define MROWS (NWIN * NTOK) // 19600
#define TOKENS (4 * 64 * 64) // 16384

// ---------------- helpers ----------------
static __device__ __forceinline__ unsigned short f2bf(float f) {
  unsigned int u = __builtin_bit_cast(unsigned int, f);
  u += 0x7fffu + ((u >> 16) & 1u);
  return (unsigned short)(u >> 16);
}
static __device__ __forceinline__ float bf2f(unsigned short h) {
  unsigned int u = ((unsigned int)h) << 16;
  return __builtin_bit_cast(float, u);
}
struct FragBits { unsigned int v[8]; };
static __device__ __forceinline__ v16bf mkfrag(uint4 a, uint4 b) {
  FragBits f;
  f.v[0] = a.x; f.v[1] = a.y; f.v[2] = a.z; f.v[3] = a.w;
  f.v[4] = b.x; f.v[5] = b.y; f.v[6] = b.z; f.v[7] = b.w;
  return __builtin_bit_cast(v16bf, f);
}

// ---------------- f32 -> bf16 weight convert ----------------
__global__ void k_f32_to_bf16(const float* __restrict__ src,
                              unsigned short* __restrict__ dst, int n) {
  int i = blockIdx.x * blockDim.x + threadIdx.x;
  int st = gridDim.x * blockDim.x;
  for (; i < n; i += st) dst[i] = f2bf(src[i]);
}

// ---------------- LayerNorm (+optional window partition) ----------------
template <int PARTITION>
__global__ __launch_bounds__(256)
void k_layernorm(const float* __restrict__ x, const float* __restrict__ w,
                 const float* __restrict__ b, unsigned short* __restrict__ out) {
  const int row = blockIdx.x;
  const int tid = threadIdx.x;
  const float* src = nullptr;
  bool valid = true;
  if (PARTITION) {
    int win = row / NTOK, n = row % NTOK;
    int bb = win / 25, rem = win % 25;
    int h = (rem / 5) * WSZ + n / WSZ, wc = (rem % 5) * WSZ + n % WSZ;
    valid = (h < 64) && (wc < 64);
    if (valid) src = x + (((size_t)bb * 64 + h) * 64 + wc) * DIM;
  } else {
    src = x + (size_t)row * DIM;
  }
  __shared__ float red[16];
  float v0 = 0.f, v1 = 0.f, v2 = 0.f;
  if (valid) { v0 = src[tid]; v1 = src[tid + 256]; v2 = src[tid + 512]; }
  float s = v0 + v1 + v2;
  float s2 = v0 * v0 + v1 * v1 + v2 * v2;
#pragma unroll
  for (int m = 1; m < 32; m <<= 1) {
    s  += __shfl_xor(s,  m, 32);
    s2 += __shfl_xor(s2, m, 32);
  }
  int wid = tid >> 5, lane = tid & 31;
  if (lane == 0) { red[wid] = s; red[8 + wid] = s2; }
  __syncthreads();
  if (tid == 0) {
    float a = 0.f, c = 0.f;
    for (int k = 0; k < 8; ++k) { a += red[k]; c += red[8 + k]; }
    red[0] = a; red[8] = c;
  }
  __syncthreads();
  float mean = red[0] * (1.0f / DIM);
  float var  = red[8] * (1.0f / DIM) - mean * mean;
  float rstd = rsqrtf(var + 1e-5f);
  unsigned short* o = out + (size_t)row * DIM;
  if (valid) {
    o[tid]       = f2bf((v0 - mean) * rstd * w[tid]       + b[tid]);
    o[tid + 256] = f2bf((v1 - mean) * rstd * w[tid + 256] + b[tid + 256]);
    o[tid + 512] = f2bf((v2 - mean) * rstd * w[tid + 512] + b[tid + 512]);
  } else {
    o[tid] = 0; o[tid + 256] = 0; o[tid + 512] = 0;
  }
}

// ---------------- WMMA GEMM, 128x128 block tile, double-buffered LDS ----------------
#define EPI_BF16      0
#define EPI_GELU_BF16 1
#define EPI_PROJ      2
#define EPI_FC2       3

template <int MODE>
__global__ __launch_bounds__(256)
void k_gemm(const unsigned short* __restrict__ A, const unsigned short* __restrict__ B,
            const float* __restrict__ bias, int M, int K, int N,
            unsigned short* __restrict__ outbf, float* __restrict__ outf,
            const float* __restrict__ resid) {
  __shared__ unsigned short sA[2][128 * 32];   // [m][k]
  __shared__ unsigned int   sB[2][128 * 16];   // Bt pair-packed [n][k/2]
  const int tid = threadIdx.x;
  const int wid = tid >> 5, lane = tid & 31;
  const int wm = wid >> 1, wn = wid & 1;       // 4 x 2 wave grid, wave tile 32M x 64N
  const int m0 = blockIdx.x * 128;
  const int n0 = blockIdx.y * 128;
  const int lh = lane >> 4, ll = lane & 15;
  // staging roles
  const int arow = tid >> 1, ahalf = tid & 1;  // A: 128 rows x 2 halves of 16
  const int bkp  = tid >> 4, bng  = tid & 15;  // B: 16 k-pairs x 16 groups of 8 n
  const int gmA = m0 + arow;
  const unsigned short* aptr = A + (size_t)gmA * K + ahalf * 16;
  const unsigned short* bptr = B + (size_t)(2 * bkp) * N + n0 + bng * 8;

  v8f acc[2][4];
#pragma unroll
  for (int i = 0; i < 2; ++i)
#pragma unroll
    for (int j = 0; j < 4; ++j)
#pragma unroll
      for (int e = 0; e < 8; ++e) acc[i][j][e] = 0.f;

  const int kTiles = K >> 5;
  uint4 a0 = {0, 0, 0, 0}, a1 = {0, 0, 0, 0}, bl, bh;
  if (gmA < M) { const uint4* g = (const uint4*)aptr; a0 = g[0]; a1 = g[1]; }
  bl = *(const uint4*)bptr;
  bh = *(const uint4*)(bptr + N);
  {   // store buffer 0
    uint4* d = (uint4*)(sA[0] + arow * 32 + ahalf * 16);
    d[0] = a0; d[1] = a1;
    union { uint4 v; unsigned short s[8]; } lo, hi;
    lo.v = bl; hi.v = bh;
#pragma unroll
    for (int j = 0; j < 8; ++j)
      sB[0][(bng * 8 + j) * 16 + bkp] = (unsigned int)lo.s[j] | ((unsigned int)hi.s[j] << 16);
  }

  for (int kt = 0; kt < kTiles; ++kt) {
    __syncthreads();
    const int cur = kt & 1;
    const bool more = (kt + 1) < kTiles;
    if (more) {  // issue next-tile global loads; latency hidden under WMMAs
      const int k0 = (kt + 1) << 5;
      a0 = make_uint4(0, 0, 0, 0); a1 = a0;
      if (gmA < M) { const uint4* g = (const uint4*)(aptr + k0); a0 = g[0]; a1 = g[1]; }
      const unsigned short* bp = bptr + (size_t)k0 * N;
      bl = *(const uint4*)bp;
      bh = *(const uint4*)(bp + N);
    }
    v16bf af[2];
#pragma unroll
    for (int mt = 0; mt < 2; ++mt) {
      const unsigned short* p = sA[cur] + (wm * 32 + mt * 16 + ll) * 32 + lh * 8;
      af[mt] = mkfrag(*(const uint4*)p, *(const uint4*)(p + 16));
    }
#pragma unroll
    for (int nt = 0; nt < 4; ++nt) {
      const unsigned int* p = sB[cur] + (wn * 64 + nt * 16 + ll) * 16 + lh * 4;
      v16bf bfr = mkfrag(*(const uint4*)p, *(const uint4*)(p + 8));
#pragma unroll
      for (int mt = 0; mt < 2; ++mt)
        acc[mt][nt] = __builtin_amdgcn_wmma_f32_16x16x32_bf16(
            false, af[mt], false, bfr, (short)0, acc[mt][nt], false, false);
    }
    __syncthreads();
    if (more) {
      uint4* d = (uint4*)(sA[cur ^ 1] + arow * 32 + ahalf * 16);
      d[0] = a0; d[1] = a1;
      union { uint4 v; unsigned short s[8]; } lo, hi;
      lo.v = bl; hi.v = bh;
#pragma unroll
      for (int j = 0; j < 8; ++j)
        sB[cur ^ 1][(bng * 8 + j) * 16 + bkp] = (unsigned int)lo.s[j] | ((unsigned int)hi.s[j] << 16);
    }
  }

  // epilogue
#pragma unroll
  for (int mt = 0; mt < 2; ++mt)
#pragma unroll
    for (int nt = 0; nt < 4; ++nt) {
      int gn = n0 + wn * 64 + nt * 16 + ll;
      float bv = bias[gn];
#pragma unroll
      for (int r = 0; r < 8; ++r) {
        int gm = m0 + wm * 32 + mt * 16 + r + 8 * lh;
        if (gm >= M) continue;
        float v = acc[mt][nt][r] + bv;
        if (MODE == EPI_BF16) {
          outbf[(size_t)gm * N + gn] = f2bf(v);
        } else if (MODE == EPI_GELU_BF16) {
          v = 0.5f * v * (1.0f + erff(v * 0.70710678118f));
          outbf[(size_t)gm * N + gn] = f2bf(v);
        } else if (MODE == EPI_PROJ) {
          int win = gm / NTOK, n = gm % NTOK;
          int bb = win / 25, rem = win % 25;
          int h = (rem / 5) * WSZ + n / WSZ;
          int wc = (rem % 5) * WSZ + n % WSZ;
          if (h < 64 && wc < 64) {
            size_t idx = (((size_t)bb * 64 + h) * 64 + wc) * DIM + gn;
            outf[idx] = v + resid[idx];
          }
        } else {  // EPI_FC2
          size_t idx = (size_t)gm * N + gn;
          outf[idx] = v + resid[idx];
        }
      }
    }
}

// ---------------- windowed attention, flash softmax, branchless hot loop ----------------
__global__ __launch_bounds__(256)
void k_attn(const unsigned short* __restrict__ qkv,
            const float* __restrict__ rel_pos_h, const float* __restrict__ rel_pos_w,
            unsigned short* __restrict__ outbf) {
  extern __shared__ __align__(16) char smem[];
  unsigned short* sQ = (unsigned short*)smem;             // [208][64]
  unsigned short* sK = sQ + 208 * 64;                     // [224][64]
  unsigned int*   sV = (unsigned int*)(sK + 224 * 64);    // Vt pair-packed [64][112]
  float* sRH = (float*)(sV + 64 * 112);                   // [196*14]
  float* sRW = sRH + NTOK * WSZ;                          // [196*14]

  const int win = blockIdx.x, head = blockIdx.y;
  const int tid = threadIdx.x;
  const size_t rs = 3 * DIM;  // 2304
  const unsigned short* qb_g = qkv + (size_t)win * NTOK * rs + head * HD;
  const unsigned short* kb_g = qb_g + DIM;
  const unsigned short* vb_g = qb_g + 2 * DIM;

  for (int idx = tid; idx < 208 * 8; idx += 256) {        // stage Q
    int row = idx >> 3, seg = idx & 7;
    uint4 z = {0, 0, 0, 0};
    if (row < NTOK) z = *(const uint4*)(qb_g + (size_t)row * rs + seg * 8);
    *(uint4*)(sQ + row * 64 + seg * 8) = z;
  }
  for (int idx = tid; idx < 224 * 8; idx += 256) {        // stage K
    int row = idx >> 3, seg = idx & 7;
    uint4 z = {0, 0, 0, 0};
    if (row < NTOK) z = *(const uint4*)(kb_g + (size_t)row * rs + seg * 8);
    *(uint4*)(sK + row * 64 + seg * 8) = z;
  }
  for (int idx = tid; idx < 112 * 8; idx += 256) {        // stage Vt (transpose, pack k-pairs)
    int kp = idx >> 3, seg = idx & 7;
    union { uint4 v; unsigned short s[8]; } a, b;
    a.v = make_uint4(0, 0, 0, 0); b.v = make_uint4(0, 0, 0, 0);
    int kr = 2 * kp;
    if (kr < NTOK)     a.v = *(const uint4*)(vb_g + (size_t)kr * rs + seg * 8);
    if (kr + 1 < NTOK) b.v = *(const uint4*)(vb_g + (size_t)(kr + 1) * rs + seg * 8);
#pragma unroll
    for (int j = 0; j < 8; ++j) {
      int d = seg * 8 + j;
      sV[d * 112 + kp] = (unsigned int)a.s[j] | ((unsigned int)b.s[j] << 16);
    }
  }
  __syncthreads();
  // decomposed rel-pos tables from (unscaled) Q
  for (int idx = tid; idx < NTOK * WSZ; idx += 256) {
    int q = idx / WSZ, kk = idx % WSZ;
    int qh = q / WSZ, qw = q % WSZ;
    const unsigned short* qr = sQ + q * 64;
    const float* rh = rel_pos_h + (size_t)(qh - kk + WSZ - 1) * HD;
    const float* rw = rel_pos_w + (size_t)(qw - kk + WSZ - 1) * HD;
    float ah = 0.f, aw = 0.f;
    for (int d = 0; d < HD; ++d) {
      float qv = bf2f(qr[d]);
      ah += qv * rh[d]; aw += qv * rw[d];
    }
    sRH[idx] = ah; sRW[idx] = aw;
  }
  __syncthreads();

  const int wid = tid >> 5, lane = tid & 31;
  const int lh = lane >> 4, ll = lane & 15;
  for (int qt = wid; qt < 13; qt += 8) {                  // 13 q-tiles over 8 waves
    v16bf qfr[2];
    {
      const unsigned short* p = sQ + (qt * 16 + ll) * 64;
      qfr[0] = mkfrag(*(const uint4*)(p + lh * 8),      *(const uint4*)(p + lh * 8 + 16));
      qfr[1] = mkfrag(*(const uint4*)(p + 32 + lh * 8), *(const uint4*)(p + 32 + lh * 8 + 16));
    }
    v8f acc[4];
#pragma unroll
    for (int dt = 0; dt < 4; ++dt)
#pragma unroll
      for (int e = 0; e < 8; ++e) acc[dt][e] = 0.f;
    float run_max = -1e30f, run_sum = 0.f;
    const int qg = qt * 16 + ll;                          // this lane's q column
    const int qsafe = (qg < NTOK) ? qg : (NTOK - 1);      // clamp: pad rows discarded later
    const float* rhrow = sRH + qsafe * WSZ;
    const float* rwrow = sRW + qsafe * WSZ;

    // ---- 6 fully-unmasked k-tile pairs (kg 0..191 all valid) ----
    for (int p2 = 0; p2 < 6; ++p2) {
      v8f s[2];
#pragma unroll
      for (int t = 0; t < 2; ++t) {
        const unsigned short* kp = sK + ((p2 * 2 + t) * 16 + ll) * 64;
        v16bf ka0 = mkfrag(*(const uint4*)(kp + lh * 8),      *(const uint4*)(kp + lh * 8 + 16));
        v16bf ka1 = mkfrag(*(const uint4*)(kp + 32 + lh * 8), *(const uint4*)(kp + 32 + lh * 8 + 16));
        v8f z;
#pragma unroll
        for (int e = 0; e < 8; ++e) z[e] = 0.f;
        z = __builtin_amdgcn_wmma_f32_16x16x32_bf16(false, ka0, false, qfr[0], (short)0, z, false, false);
        z = __builtin_amdgcn_wmma_f32_16x16x32_bf16(false, ka1, false, qfr[1], (short)0, z, false, false);
        s[t] = z;
      }
      float m_loc = -1e30f;
#pragma unroll
      for (int t = 0; t < 2; ++t) {
        int kg0 = (p2 * 2 + t) * 16 + 8 * lh;
        int kh = kg0 / WSZ, kw = kg0 % WSZ;               // once per tile, then carry
#pragma unroll
        for (int r = 0; r < 8; ++r) {
          float v = s[t][r] * 0.125f + rhrow[kh] + rwrow[kw];
          s[t][r] = v;
          m_loc = fmaxf(m_loc, v);
          int c = (kw == WSZ - 1);
          kh += c;
          kw = c ? 0 : kw + 1;
        }
      }
      m_loc = fmaxf(m_loc, __shfl_xor(m_loc, 16, 32));
      float m_new = fmaxf(run_max, m_loc);
      float alpha = __expf(run_max - m_new);
      run_max = m_new;
      float f[8];
#pragma unroll
      for (int r = 0; r < 8; ++r) f[r] = __shfl(alpha, r + 8 * lh, 32);
#pragma unroll
      for (int dt = 0; dt < 4; ++dt)
#pragma unroll
        for (int r = 0; r < 8; ++r) acc[dt][r] *= f[r];
      run_sum *= alpha;
      union { uint4 v[2]; unsigned short s[16]; } pe;
#pragma unroll
      for (int t = 0; t < 2; ++t)
#pragma unroll
        for (int r = 0; r < 8; ++r) {
          float e = __expf(s[t][r] - m_new);
          run_sum += e;
          pe.s[t * 8 + r] = f2bf(e);                      // C-layout pair == A-frag layout
        }
      v16bf pf = mkfrag(pe.v[0], pe.v[1]);
#pragma unroll
      for (int dt = 0; dt < 4; ++dt) {
        const unsigned int* vp = sV + (dt * 16 + ll) * 112 + p2 * 16 + lh * 4;
        v16bf vb = mkfrag(*(const uint4*)vp, *(const uint4*)(vp + 8));
        acc[dt] = __builtin_amdgcn_wmma_f32_16x16x32_bf16(false, pf, false, vb, (short)0, acc[dt], false, false);
      }
    }

    // ---- masked tail: single k-tile 12 (kg 192..207; only 192..195 valid) ----
    {
      const unsigned short* kp = sK + (12 * 16 + ll) * 64;
      v16bf ka0 = mkfrag(*(const uint4*)(kp + lh * 8),      *(const uint4*)(kp + lh * 8 + 16));
      v16bf ka1 = mkfrag(*(const uint4*)(kp + 32 + lh * 8), *(const uint4*)(kp + 32 + lh * 8 + 16));
      v8f s0;
#pragma unroll
      for (int e = 0; e < 8; ++e) s0[e] = 0.f;
      s0 = __builtin_amdgcn_wmma_f32_16x16x32_bf16(false, ka0, false, qfr[0], (short)0, s0, false, false);
      s0 = __builtin_amdgcn_wmma_f32_16x16x32_bf16(false, ka1, false, qfr[1], (short)0, s0, false, false);
      const float rh13 = rhrow[13];
      float m_loc = -1e30f;
#pragma unroll
      for (int r = 0; r < 8; ++r) {
        int j = r + 8 * lh;                               // kg = 192 + j, kh = 13, kw = 10 + j
        int kwc = (j + 10 > 13) ? 13 : (j + 10);
        float bias = rh13 + rwrow[kwc];
        float v = (j < 4) ? s0[r] * 0.125f + bias : -1e30f;
        s0[r] = v;
        m_loc = fmaxf(m_loc, v);
      }
      m_loc = fmaxf(m_loc, __shfl_xor(m_loc, 16, 32));
      float m_new = fmaxf(run_max, m_loc);
      float alpha = __expf(run_max - m_new);
      run_max = m_new;
      float f[8];
#pragma unroll
      for (int r = 0; r < 8; ++r) f[r] = __shfl(alpha, r + 8 * lh, 32);
#pragma unroll
      for (int dt = 0; dt < 4; ++dt)
#pragma unroll
        for (int r = 0; r < 8; ++r) acc[dt][r] *= f[r];
      run_sum *= alpha;
      union { uint4 v[2]; unsigned short s[16]; } pe;
#pragma unroll
      for (int r = 0; r < 8; ++r) {
        float e = __expf(s0[r] - m_new);
        run_sum += e;
        pe.s[r] = f2bf(e);
      }
      pe.v[1] = make_uint4(0, 0, 0, 0);                   // k-tile 13 is pure padding
      v16bf pf = mkfrag(pe.v[0], pe.v[1]);
#pragma unroll
      for (int dt = 0; dt < 4; ++dt) {
        const unsigned int* vp = sV + (dt * 16 + ll) * 112 + 6 * 16 + lh * 4;
        v16bf vb = mkfrag(*(const uint4*)vp, *(const uint4*)(vp + 8));
        acc[dt] = __builtin_amdgcn_wmma_f32_16x16x32_bf16(false, pf, false, vb, (short)0, acc[dt], false, false);
      }
    }

    float tot = run_sum + __shfl_xor(run_sum, 16, 32);
    float inv = 1.0f / tot;
    float g[8];
#pragma unroll
    for (int r = 0; r < 8; ++r) g[r] = __shfl(inv, r + 8 * lh, 32);
#pragma unroll
    for (int dt = 0; dt < 4; ++dt)
#pragma unroll
      for (int r = 0; r < 8; ++r) {
        int q = qt * 16 + r + 8 * lh;
        if (q >= NTOK) continue;
        outbf[((size_t)win * NTOK + q) * DIM + head * HD + dt * 16 + ll] = f2bf(acc[dt][r] * g[r]);
      }
  }
}

// ---------------- launcher ----------------
extern "C" void kernel_launch(void* const* d_in, const int* in_sizes, int n_in,
                              void* d_out, int out_size, void* d_ws, size_t ws_size,
                              hipStream_t stream) {
  (void)in_sizes; (void)n_in; (void)out_size; (void)ws_size;
  const float* x      = (const float*)d_in[0];
  const float* ln1_w  = (const float*)d_in[1];
  const float* ln1_b  = (const float*)d_in[2];
  const float* qkv_w  = (const float*)d_in[3];
  const float* qkv_b  = (const float*)d_in[4];
  const float* proj_w = (const float*)d_in[5];
  const float* proj_b = (const float*)d_in[6];
  const float* rph    = (const float*)d_in[7];
  const float* rpw    = (const float*)d_in[8];
  const float* ln2_w  = (const float*)d_in[9];
  const float* ln2_b  = (const float*)d_in[10];
  const float* fc1_w  = (const float*)d_in[11];
  const float* fc1_b  = (const float*)d_in[12];
  const float* fc2_w  = (const float*)d_in[13];
  const float* fc2_b  = (const float*)d_in[14];
  float* out = (float*)d_out;

  char* ws = (char*)d_ws;
  size_t off = 0;
  auto alloc = [&](size_t bytes) -> char* {
    char* p = ws + off;
    off = (off + bytes + 255) & ~(size_t)255;
    return p;
  };
  unsigned short* wq = (unsigned short*)alloc((size_t)DIM * 3 * DIM * 2);
  unsigned short* wp = (unsigned short*)alloc((size_t)DIM * DIM * 2);
  unsigned short* w1 = (unsigned short*)alloc((size_t)DIM * 4 * DIM * 2);
  unsigned short* w2 = (unsigned short*)alloc((size_t)4 * DIM * DIM * 2);
  unsigned short* xw = (unsigned short*)alloc((size_t)MROWS * DIM * 2);       // LN1 rows, reused for attn out
  unsigned short* qh = (unsigned short*)alloc((size_t)TOKENS * 4 * DIM * 2);  // QKV / MLP hidden (shared, max size)
  float*          y  = (float*)alloc((size_t)TOKENS * DIM * 4);               // post-attention residual
  unsigned short* l2 = (unsigned short*)alloc((size_t)TOKENS * DIM * 2);      // LN2 rows

  k_f32_to_bf16<<<1024, 256, 0, stream>>>(qkv_w,  wq, DIM * 3 * DIM);
  k_f32_to_bf16<<<1024, 256, 0, stream>>>(proj_w, wp, DIM * DIM);
  k_f32_to_bf16<<<1024, 256, 0, stream>>>(fc1_w,  w1, DIM * 4 * DIM);
  k_f32_to_bf16<<<1024, 256, 0, stream>>>(fc2_w,  w2, 4 * DIM * DIM);

  k_layernorm<1><<<MROWS, 256, 0, stream>>>(x, ln1_w, ln1_b, xw);

  dim3 gq((MROWS + 127) / 128, (3 * DIM) / 128);
  k_gemm<EPI_BF16><<<gq, 256, 0, stream>>>(xw, wq, qkv_b, MROWS, DIM, 3 * DIM, qh, nullptr, nullptr);

  size_t smem = (size_t)(208 * 64 + 224 * 64) * 2 + (size_t)64 * 112 * 4 + (size_t)2 * NTOK * WSZ * 4;
  dim3 ga(NWIN, NHEAD);
  k_attn<<<ga, 256, smem, stream>>>(qh, rph, rpw, xw);

  dim3 gp((MROWS + 127) / 128, DIM / 128);
  k_gemm<EPI_PROJ><<<gp, 256, 0, stream>>>(xw, wp, proj_b, MROWS, DIM, DIM, nullptr, y, x);

  k_layernorm<0><<<TOKENS, 256, 0, stream>>>(y, ln2_w, ln2_b, l2);

  dim3 g1(TOKENS / 128, (4 * DIM) / 128);
  k_gemm<EPI_GELU_BF16><<<g1, 256, 0, stream>>>(l2, w1, fc1_b, TOKENS, DIM, 4 * DIM, qh, nullptr, nullptr);

  dim3 g2(TOKENS / 128, DIM / 128);
  k_gemm<EPI_FC2><<<g2, 256, 0, stream>>>(qh, w2, fc2_b, TOKENS, 4 * DIM, DIM, nullptr, out, y);
}
